// VehicleActor_33698313404986
// MI455X (gfx1250) — matrix operand order repaired and checked by reading, not compiled, not run on hardware
//
#include <hip/hip_runtime.h>
#include <hip/hip_bf16.h>

// ---------------------------------------------------------------------------
// VehicleActor: LayerNorm -> 2-layer LSTM (T=128) -> tanh FC head
// CDNA5 (gfx1250) wave32 WMMA implementation.
// Round 5: native v_tanh_f32 (gfx1250 TRANS op) for the LSTM activations via
// __builtin_amdgcn_tanhf (guarded), halving transcendental pressure in the
// cell updates. GEMM core unchanged from round 4 (immediate-offset B loads,
// 8-deep fragment buffering, BM=32 with 2x B-fragment reuse).
// ---------------------------------------------------------------------------

#define BATCH 1024
#define TSEQ  128
#define DIN   64
#define HID   256
#define GATE  1024      // 4*HID
#define ADIM  8
#define BM    32        // batch rows per workgroup (two WMMA M tiles)
#define NTILES 64       // GATE / 16
#define KT0   10        // (DIN + HID) / 32  : layer0 fused [xn|h0] GEMM
#define KT1   16        // (HID + HID) / 32  : layer1 fused [h0|h1] GEMM
#define HSTRIDE 584     // padded hcat row stride (halves): 1168B, conflict-free

typedef __attribute__((ext_vector_type(16))) _Float16 v16h;
typedef __attribute__((ext_vector_type(8)))  _Float16 v8h;
typedef __attribute__((ext_vector_type(8)))  float    v8f;

#if __has_builtin(__builtin_amdgcn_tanhf)
static __device__ __forceinline__ float tanh_f(float x) {
    return __builtin_amdgcn_tanhf(x);          // native v_tanh_f32 (1 TRANS op)
}
static __device__ __forceinline__ float sigmoid_f(float x) {
    return fmaf(0.5f, __builtin_amdgcn_tanhf(0.5f * x), 0.5f);
}
#else
static __device__ __forceinline__ float tanh_f(float x) {
    float e = __expf(2.0f * x);
    return (e - 1.0f) / (e + 1.0f);
}
static __device__ __forceinline__ float sigmoid_f(float x) {
    return 1.0f / (1.0f + __expf(-x));
}
#endif

// A-fragment load: 16-bit A matrix 16x32, contiguous k order (matches pack_w):
//   lane<16 : M=lane,    K = {0..15}
//   lane>=16: M=lane-16, K = {16..31}
static __device__ __forceinline__ v16h load_a(const _Float16* rp, int hi) {
    v8h lo = *(const v8h*)(rp + hi * 16);
    v8h up = *(const v8h*)(rp + hi * 16 + 8);
    return __builtin_shufflevector(lo, up, 0,1,2,3,4,5,6,7,8,9,10,11,12,13,14,15);
}

static __device__ __forceinline__ v8f wmma_f16(v16h a, v16h b, v8f c) {
    return __builtin_amdgcn_wmma_f32_16x16x32_f16(false, a, false, b, (short)0, c,
                                                  false, false);
}

// ---------------------------------------------------------------------------
// Prep: LayerNorm -> f16 (one wave per batch row; 2 elems/lane, shfl reduce)
// ---------------------------------------------------------------------------
__global__ void ln_kernel(const float* __restrict__ x, const float* __restrict__ gamma,
                          const float* __restrict__ beta, _Float16* __restrict__ xnh) {
    int wave = (blockIdx.x * blockDim.x + threadIdx.x) >> 5;
    int lane = threadIdx.x & 31;
    if (wave >= BATCH) return;
    const float* row = x + (size_t)wave * DIN;
    float a = row[lane], b = row[lane + 32];
    float s = a + b;
    for (int off = 16; off; off >>= 1) s += __shfl_xor(s, off, 32);
    float mu = s * (1.0f / (float)DIN);
    float da = a - mu, db = b - mu;
    float v = da * da + db * db;
    for (int off = 16; off; off >>= 1) v += __shfl_xor(v, off, 32);
    float rstd = rsqrtf(v * (1.0f / (float)DIN) + 1e-5f);
    xnh[(size_t)wave * DIN + lane]      = (_Float16)(da * rstd * gamma[lane] + beta[lane]);
    xnh[(size_t)wave * DIN + lane + 32] = (_Float16)(db * rstd * gamma[lane + 32] + beta[lane + 32]);
}

// ---------------------------------------------------------------------------
// Prep: pack W (GATE x Ksrc, row-major f32) into WMMA B-fragment order (f16).
// Packed element layout: [nt][ktOff+ktl][lane][j] with
//   value = W[nt*16 + (lane&15)][ktl*32 + (lane>>4)*16 + j]
// so a B fragment is a single contiguous 32B load per lane.
// ---------------------------------------------------------------------------
__global__ void pack_w(const float* __restrict__ W, int Ksrc, int ktOff, int ktTot,
                       _Float16* __restrict__ dst) {
    int t    = blockIdx.x * blockDim.x + threadIdx.x;
    int lane = t & 31;
    int blk  = t >> 5;
    int ktCnt = Ksrc / 32;
    if (blk >= NTILES * ktCnt) return;
    int nt  = blk / ktCnt;
    int ktl = blk % ktCnt;
    int n = nt * 16 + (lane & 15);
    int k = ktl * 32 + (lane >> 4) * 16;
    const float* src = W + (size_t)n * Ksrc + k;
    _Float16* d = dst + (((size_t)nt * ktTot + (ktOff + ktl)) * 32 + lane) * 16;
#pragma unroll
    for (int j = 0; j < 16; ++j) d[j] = (_Float16)src[j];
}

__global__ void add_bias(const float* __restrict__ a, const float* __restrict__ b,
                         float* __restrict__ o, int n) {
    int i = blockIdx.x * blockDim.x + threadIdx.x;
    if (i < n) o[i] = a[i] + b[i];
}

// ---------------------------------------------------------------------------
// Persistent LSTM kernel: 32 workgroups x 256 threads (8 waves).
// Each WG owns 32 batch rows for all 128 steps. Per kt: 16 global_load_b128
// fill bb[0..7] (one B fragment per gate tile), then 16 WMMAs consume them
// (two M tiles per fragment). hcat = [xn | h0 | h1] f16; layer-0 GEMM K=320
// over [xn|h0], layer-1 K=512 over [h0|h1]. Wave w owns gate tiles {w+8j} so
// i,f,g,o pair up in-wave and the cell update is register-local.
// ---------------------------------------------------------------------------
__global__ __launch_bounds__(256, 1)
void lstm_persist(const _Float16* __restrict__ xnh,
                  const _Float16* __restrict__ pk0,   // stacked [W_ih0;W_hh0], K=320
                  const _Float16* __restrict__ pk1,   // stacked [W_ih1;W_hh1], K=512
                  const float* __restrict__ b0g,
                  const float* __restrict__ b1g,
                  const float* __restrict__ fcW,
                  const float* __restrict__ fcb,
                  float* __restrict__ out) {
    __shared__ _Float16 hcat[BM * HSTRIDE];   // 36.5KB [m][0:64]=xn|[64:320]=h0|[320:576]=h1
    __shared__ float    c0s[BM * HID];        // 32KB
    __shared__ float    c1s[BM * HID];        // 32KB
    __shared__ float    fcWs[ADIM * HID];     // 8KB
    __shared__ float    b0s[GATE];            // 4KB
    __shared__ float    b1s[GATE];            // 4KB
    __shared__ float    fcbs[ADIM];

    const int tid  = threadIdx.x;
    const int wave = tid >> 5;
    const int lane = tid & 31;
    const int l15  = lane & 15;
    const int hi   = lane >> 4;
    const int m0   = blockIdx.x * BM;
    const int colA = wave * 16 + l15;     // gate column for even-j accumulators

    for (int i = tid; i < BM * HSTRIDE; i += 256) hcat[i] = (_Float16)0.0f;
    for (int i = tid; i < BM * HID; i += 256) { c0s[i] = 0.0f; c1s[i] = 0.0f; }
    for (int i = tid; i < ADIM * HID; i += 256) fcWs[i] = fcW[i];
    for (int i = tid; i < GATE; i += 256) { b0s[i] = b0g[i]; b1s[i] = b1g[i]; }
    if (tid < ADIM) fcbs[tid] = fcb[tid];
    __syncthreads();
    // write xn into cols [0,64) of each row (time-invariant layer-0 input)
    for (int i = tid; i < BM * DIN; i += 256) {
        int m = i >> 6, k = i & 63;
        hcat[m * HSTRIDE + k] = xnh[(size_t)(m0 + m) * DIN + k];
    }
    __syncthreads();

    // per-wave packed-weight bases: fragment (j, kt) at base + (j*8*KT + kt)*512 halves
    const _Float16* pk0w = pk0 + ((size_t)wave * KT0 * 32 + lane) * 16;
    const _Float16* pk1w = pk1 + ((size_t)wave * KT1 * 32 + lane) * 16;

    v8f acc0[8], acc1[8];   // M tiles 0 (rows 0-15) and 1 (rows 16-31)
    for (int t = 0; t < TSEQ; ++t) {
        // ========== layer 0: g = b0 + [xn|h0] @ [W_ih0;W_hh0]^T  (K=320) =====
#pragma unroll
        for (int j = 0; j < 8; ++j) {
            float bj = b0s[colA + 128 * j];
#pragma unroll
            for (int r = 0; r < 8; ++r) { acc0[j][r] = bj; acc1[j][r] = bj; }
        }
#pragma unroll 1
        for (int kt = 0; kt < KT0; ++kt) {
            v16h bb[8];
#pragma unroll
            for (int j = 0; j < 8; ++j)
                bb[j] = *(const v16h*)(pk0w + ((size_t)(j * 8 * KT0 + kt)) * 512);
            v16h a0 = load_a(&hcat[l15 * HSTRIDE + kt * 32], hi);
            v16h a1 = load_a(&hcat[(l15 + 16) * HSTRIDE + kt * 32], hi);
#pragma unroll
            for (int j = 0; j < 8; ++j) {
                acc0[j] = wmma_f16(a0, bb[j], acc0[j]);
                acc1[j] = wmma_f16(a1, bb[j], acc1[j]);
            }
        }
        __syncthreads();
        // layer 0 cell update (register-local gates, c in LDS, h -> hcat f16)
#pragma unroll
        for (int mh = 0; mh < 2; ++mh) {
#pragma unroll
            for (int p = 0; p < 2; ++p) {
                int col = colA + 128 * p;
                v8f gi = mh ? acc1[0 + p] : acc0[0 + p];
                v8f gf = mh ? acc1[2 + p] : acc0[2 + p];
                v8f gg = mh ? acc1[4 + p] : acc0[4 + p];
                v8f go = mh ? acc1[6 + p] : acc0[6 + p];
#pragma unroll
                for (int r = 0; r < 8; ++r) {
                    int m = r + 8 * hi + 16 * mh;
                    float c  = c0s[m * HID + col];
                    float cn = sigmoid_f(gf[r]) * c + sigmoid_f(gi[r]) * tanh_f(gg[r]);
                    float hn = sigmoid_f(go[r]) * tanh_f(cn);
                    c0s[m * HID + col] = cn;
                    hcat[m * HSTRIDE + DIN + col] = (_Float16)hn;
                }
            }
        }
        __syncthreads();

        // ========== layer 1: g = b1 + [h0|h1] @ [W_ih1;W_hh1]^T  (K=512) =====
#pragma unroll
        for (int j = 0; j < 8; ++j) {
            float bj = b1s[colA + 128 * j];
#pragma unroll
            for (int r = 0; r < 8; ++r) { acc0[j][r] = bj; acc1[j][r] = bj; }
        }
#pragma unroll 1
        for (int kt = 0; kt < KT1; ++kt) {
            v16h bb[8];
#pragma unroll
            for (int j = 0; j < 8; ++j)
                bb[j] = *(const v16h*)(pk1w + ((size_t)(j * 8 * KT1 + kt)) * 512);
            v16h a0 = load_a(&hcat[l15 * HSTRIDE + DIN + kt * 32], hi);
            v16h a1 = load_a(&hcat[(l15 + 16) * HSTRIDE + DIN + kt * 32], hi);
#pragma unroll
            for (int j = 0; j < 8; ++j) {
                acc0[j] = wmma_f16(a0, bb[j], acc0[j]);
                acc1[j] = wmma_f16(a1, bb[j], acc1[j]);
            }
        }
        __syncthreads();
        // layer 1 cell update
#pragma unroll
        for (int mh = 0; mh < 2; ++mh) {
#pragma unroll
            for (int p = 0; p < 2; ++p) {
                int col = colA + 128 * p;
                v8f gi = mh ? acc1[0 + p] : acc0[0 + p];
                v8f gf = mh ? acc1[2 + p] : acc0[2 + p];
                v8f gg = mh ? acc1[4 + p] : acc0[4 + p];
                v8f go = mh ? acc1[6 + p] : acc0[6 + p];
#pragma unroll
                for (int r = 0; r < 8; ++r) {
                    int m = r + 8 * hi + 16 * mh;
                    float c  = c1s[m * HID + col];
                    float cn = sigmoid_f(gf[r]) * c + sigmoid_f(gi[r]) * tanh_f(gg[r]);
                    float hn = sigmoid_f(go[r]) * tanh_f(cn);
                    c1s[m * HID + col] = cn;
                    hcat[m * HSTRIDE + DIN + HID + col] = (_Float16)hn;
                }
            }
        }
        __syncthreads();

        // ========== FC head: out[b,t,:] = tanh(h1 @ fcW^T + fcb) =============
        {
            int m = tid >> 3, aI = tid & 7;       // 256 threads = 32 rows x 8 outs
            float s = fcbs[aI];
            const _Float16* hr = &hcat[m * HSTRIDE + DIN + HID];
            const float*    wr = &fcWs[aI * HID];
#pragma unroll 8
            for (int n = 0; n < HID; ++n) s += (float)hr[n] * wr[n];
            out[((size_t)(m0 + m) * TSEQ + t) * ADIM + aI] = tanh_f(s);
        }
        // FC reads h1; next write of h1 is three barriers away -> safe
    }
}

// ---------------------------------------------------------------------------
extern "C" void kernel_launch(void* const* d_in, const int* in_sizes, int n_in,
                              void* d_out, int out_size, void* d_ws, size_t ws_size,
                              hipStream_t stream) {
    (void)in_sizes; (void)n_in; (void)out_size; (void)ws_size;
    const float* state = (const float*)d_in[0];
    const float* gamma = (const float*)d_in[1];
    const float* beta  = (const float*)d_in[2];
    const float* W_ih0 = (const float*)d_in[3];
    const float* W_hh0 = (const float*)d_in[4];
    const float* b_ih0 = (const float*)d_in[5];
    const float* b_hh0 = (const float*)d_in[6];
    const float* W_ih1 = (const float*)d_in[7];
    const float* W_hh1 = (const float*)d_in[8];
    const float* b_ih1 = (const float*)d_in[9];
    const float* b_hh1 = (const float*)d_in[10];
    const float* fcW   = (const float*)d_in[11];
    const float* fcb   = (const float*)d_in[12];

    // workspace carve-out (~1.8 MB), 256B aligned sub-buffers
    char* ws = (char*)d_ws;
    size_t off = 0;
    auto carve = [&](size_t bytes) -> char* {
        char* p = ws + off;
        off = (off + bytes + 255) & ~(size_t)255;
        return p;
    };
    _Float16* xnh = (_Float16*)carve((size_t)BATCH * DIN * 2);
    _Float16* pk0 = (_Float16*)carve((size_t)(DIN + HID) * GATE * 2);  // K=320 stack
    _Float16* pk1 = (_Float16*)carve((size_t)(HID + HID) * GATE * 2);  // K=512 stack
    float*    b0  = (float*)carve((size_t)GATE * 4);
    float*    b1  = (float*)carve((size_t)GATE * 4);

    ln_kernel<<<BATCH / 8, 256, 0, stream>>>(state, gamma, beta, xnh);
    // pk0 = [W_ih0 (kt 0..1) ; W_hh0 (kt 2..9)], ktTot = 10
    pack_w<<<(NTILES * 2 * 32 + 255) / 256, 256, 0, stream>>>(W_ih0, DIN, 0, KT0, pk0);
    pack_w<<<(NTILES * 8 * 32 + 255) / 256, 256, 0, stream>>>(W_hh0, HID, 2, KT0, pk0);
    // pk1 = [W_ih1 (kt 0..7) ; W_hh1 (kt 8..15)], ktTot = 16
    pack_w<<<(NTILES * 8 * 32 + 255) / 256, 256, 0, stream>>>(W_ih1, HID, 0, KT1, pk1);
    pack_w<<<(NTILES * 8 * 32 + 255) / 256, 256, 0, stream>>>(W_hh1, HID, 8, KT1, pk1);
    add_bias<<<(GATE + 255) / 256, 256, 0, stream>>>(b_ih0, b_hh0, b0, GATE);
    add_bias<<<(GATE + 255) / 256, 256, 0, stream>>>(b_ih1, b_hh1, b1, GATE);

    lstm_persist<<<BATCH / BM, 256, 0, stream>>>(xnh, pk0, pk1, b0, b1,
                                                 fcW, fcb, (float*)d_out);
}
